// SEGNNMessagePassing_30915174596963
// MI455X (gfx1250) — matrix operand order repaired and checked by reading, not compiled, not run on hardware
//
#include <hip/hip_runtime.h>
#include <hip/hip_bf16.h>

#define NS 64
#define NV 32
#define AC 16
#define RB 8
#define HB 8
#define GS 96
#define NW 192

typedef __attribute__((ext_vector_type(16))) __bf16 bf16x16;
typedef __attribute__((ext_vector_type(8)))  float  f32x8;

static __device__ __forceinline__ float sigm(float x){ return 1.0f/(1.0f+__expf(-x)); }
static __device__ __forceinline__ float silu(float x){ return x*sigm(x); }

#define WMMA_BF16(A,B,C) __builtin_amdgcn_wmma_f32_16x16x32_bf16(false,(A),false,(B),(short)0,(C),false,false)

// A fragment loader: tile in LDS, row-major [M rows][96 cols] bf16.
// 16-bit A 16x32 layout: lane<16 -> M=lane, K base = kt*32 + (lane>>4)*8;
// vector elems 0..7 -> K=kb..kb+7, elems 8..15 -> K=kb+16..kb+23.
static __device__ __forceinline__ bf16x16 frag_a96(const __bf16* tile, int rowOff, int kt, int lane){
  const int half = lane >> 4;
  const __bf16* row = tile + (size_t)(rowOff + (lane & 15)) * 96 + kt*32 + half*8;
  bf16x16 a;
#pragma unroll
  for (int j = 0; j < 8; ++j) a[j] = row[j];
#pragma unroll
  for (int j = 0; j < 8; ++j) a[8+j] = row[16+j];
  return a;
}

// B fragment loader: weights stored n-major [n][K] bf16 in LDS.
// 16-bit B 32x16: N=lane&15, K base = (lane>>4)*8; elems 0..7 -> K=kb..kb+7, elems 8..15 -> K=kb+16..kb+23.
static __device__ __forceinline__ bf16x16 frag_b(const __bf16* bt, int ldk, int nt, int kt, int lane){
  const __bf16* p = bt + (size_t)(nt*16 + (lane & 15)) * ldk + kt*32 + (lane >> 4)*8;
  bf16x16 b;
#pragma unroll
  for (int j = 0; j < 16; ++j) b[j] = p[(j < 8) ? j : (j + 8)];
  return b;
}

// ---------------- K1: node linears s1 = node_s@W1s/8, v1 = node_v x W1v /sqrt(32) ----
__global__ void k_node_prep(const float* __restrict__ node_s, const float* __restrict__ node_v,
                            const float* __restrict__ W1s, const float* __restrict__ W1v,
                            float* __restrict__ s1, float* __restrict__ v1, int N){
  const int t = threadIdx.x & 63;
  const int n = blockIdx.x*4 + (threadIdx.x >> 6);
  if (n >= N) return;
  const float* sr = node_s + (size_t)n*NS;
  float acc = 0.f;
#pragma unroll 8
  for (int u = 0; u < NS; ++u) acc += sr[u] * W1s[u*NS + t];
  s1[(size_t)n*NS + t] = acc * 0.125f;
  if (t < NV){
    const float* vr = node_v + (size_t)n*(NV*3);
    float a0=0.f, a1=0.f, a2=0.f;
#pragma unroll 8
    for (int u = 0; u < NV; ++u){
      float w = W1v[u*NV + t];
      a0 += vr[u*3+0]*w; a1 += vr[u*3+1]*w; a2 += vr[u*3+2]*w;
    }
    float* o = v1 + (size_t)n*(NV*3) + t*3;
    const float s = 0.17677669529663689f; // 1/sqrt(32)
    o[0]=a0*s; o[1]=a1*s; o[2]=a2*s;
  }
}

// ---------------- K2: fused edge kernel (gather + radial MLP + TP + WMMA + gate + scatter)
__global__ __launch_bounds__(64) void k_edge(
    const float* __restrict__ s1, const float* __restrict__ v1,
    const float* __restrict__ edge_emb, const float* __restrict__ edge_sh0,
    const float* __restrict__ edge_sh1, const int* __restrict__ eidx,
    const float* __restrict__ Wm1, const float* __restrict__ Wm2,
    const float* __restrict__ W2s, const float* __restrict__ W2v,
    float* __restrict__ aggs, float* __restrict__ aggv, int E)
{
  __shared__ float  sWm1[RB*HB];
  __shared__ float  sWm2[HB*NW];
  __shared__ __bf16 sW2s[GS*GS];     // transposed: [n][k]
  __shared__ __bf16 sW2v[NV*GS];     // transposed: [n][k]
  __shared__ __bf16 sAs[2][16*GS];   // per-wave tp_s tile  [16 edges][96]
  __shared__ __bf16 sAv[2][48*GS];   // per-wave tp_v tile  [3 comps * 16 edges][96]

  const int tid = threadIdx.x;
  for (int i = tid; i < RB*HB; i += 64) sWm1[i] = Wm1[i];
  for (int i = tid; i < HB*NW; i += 64) sWm2[i] = Wm2[i];
  for (int i = tid; i < GS*GS; i += 64){ int v = i/GS, u = i - v*GS; sW2s[v*GS+u] = (__bf16)W2s[u*GS + v]; }
  for (int i = tid; i < NV*GS; i += 64){ int v = i/GS, u = i - v*GS; sW2v[v*GS+u] = (__bf16)W2v[u*NV + v]; }
  __syncthreads();

  const int w    = tid >> 5;
  const int lane = tid & 31;
  const int tile = blockIdx.x*2 + w;
  const int base = tile*16;
  const bool tv  = (base + 16) <= E;
  __bf16* As = sAs[w];
  __bf16* Av = sAv[w];

  if (tv){
    const int e = lane & 15, h = lane >> 4;
    const int ge = base + e;
    const int src = eidx[ge];
    float emb[RB];
#pragma unroll
    for (int i = 0; i < RB; ++i) emb[i] = edge_emb[ge*RB + i];
    float hv[HB];
#pragma unroll
    for (int j = 0; j < HB; ++j){
      float a = 0.f;
#pragma unroll
      for (int i = 0; i < RB; ++i) a += emb[i]*sWm1[i*HB + j];
      hv[j] = silu(a);
    }
    const float sh0 = edge_sh0[ge];
    float sh1v[3];
#pragma unroll
    for (int c = 0; c < 3; ++c) sh1v[c] = edge_sh1[ge*3 + c];
    const float* ssp = s1 + (size_t)src*NS;
    const float* vvp = v1 + (size_t)src*(NV*3);

    if (h == 0){
      // w[0..63]=w0a -> tp_s cols 0..63 ; w[64..95]=w1a[0..31] -> tp_v cols 0..31
      for (int u = 0; u < NS; ++u){
        float wv = 0.f;
#pragma unroll
        for (int j = 0; j < HB; ++j) wv += hv[j]*sWm2[j*NW + u];
        As[e*GS + u] = (__bf16)(wv * ssp[u] * sh0);
      }
      for (int u = 0; u < 32; ++u){
        float wv = 0.f;
#pragma unroll
        for (int j = 0; j < HB; ++j) wv += hv[j]*sWm2[j*NW + NS + u];
        const float s = ssp[u];
#pragma unroll
        for (int c = 0; c < 3; ++c) Av[(c*16+e)*GS + u] = (__bf16)(wv * s * sh1v[c]);
      }
    } else {
      // w[96..127]=w1a[32..63] ; w[128..159]=w1b ; w[160..191]=w0b
      for (int u = 32; u < NS; ++u){
        float wv = 0.f;
#pragma unroll
        for (int j = 0; j < HB; ++j) wv += hv[j]*sWm2[j*NW + NS + u];
        const float s = ssp[u];
#pragma unroll
        for (int c = 0; c < 3; ++c) Av[(c*16+e)*GS + u] = (__bf16)(wv * s * sh1v[c]);
      }
      for (int jv = 0; jv < NV; ++jv){
        float wv = 0.f;
#pragma unroll
        for (int j = 0; j < HB; ++j) wv += hv[j]*sWm2[j*NW + 128 + jv];
#pragma unroll
        for (int c = 0; c < 3; ++c) Av[(c*16+e)*GS + NS + jv] = (__bf16)(wv * vvp[jv*3+c] * sh0);
      }
      for (int jv = 0; jv < NV; ++jv){
        float wv = 0.f;
#pragma unroll
        for (int j = 0; j < HB; ++j) wv += hv[j]*sWm2[j*NW + 160 + jv];
        float dot = 0.f;
#pragma unroll
        for (int c = 0; c < 3; ++c) dot += vvp[jv*3+c]*sh1v[c];
        As[e*GS + NS + jv] = (__bf16)(wv * dot);
      }
    }
  }
  __syncthreads();

  if (tv){
    f32x8 accS[6] = {};
    f32x8 accV[6] = {};
    bf16x16 a0 = frag_a96(As, 0, 0, lane);
    bf16x16 a1 = frag_a96(As, 0, 1, lane);
    bf16x16 a2 = frag_a96(As, 0, 2, lane);
#pragma unroll
    for (int nt = 0; nt < 6; ++nt){
      accS[nt] = WMMA_BF16(a0, frag_b(sW2s, GS, nt, 0, lane), accS[nt]);
      accS[nt] = WMMA_BF16(a1, frag_b(sW2s, GS, nt, 1, lane), accS[nt]);
      accS[nt] = WMMA_BF16(a2, frag_b(sW2s, GS, nt, 2, lane), accS[nt]);
    }
#pragma unroll
    for (int mt = 0; mt < 3; ++mt){
      bf16x16 va0 = frag_a96(Av, mt*16, 0, lane);
      bf16x16 va1 = frag_a96(Av, mt*16, 1, lane);
      bf16x16 va2 = frag_a96(Av, mt*16, 2, lane);
#pragma unroll
      for (int nt = 0; nt < 2; ++nt){
        const int ci = mt*2 + nt;
        accV[ci] = WMMA_BF16(va0, frag_b(sW2v, GS, nt, 0, lane), accV[ci]);
        accV[ci] = WMMA_BF16(va1, frag_b(sW2v, GS, nt, 1, lane), accV[ci]);
        accV[ci] = WMMA_BF16(va2, frag_b(sW2v, GS, nt, 2, lane), accV[ci]);
      }
    }
    // gate + scatter; gate channels sit at the same (lane, r) slot as the vector channels
    const float inv96  = 0.10206207261596575f;   // 1/sqrt(96)
    const float invavg = 0.28867513459481287f;   // 1/sqrt(12)
    const int lc = lane & 15, half = lane >> 4;
#pragma unroll
    for (int r = 0; r < 8; ++r){
      const int e = r + 8*half;
      const int d = eidx[E + base + e];
      const float g0 = sigm(accS[4][r]*inv96);
      const float g1 = sigm(accS[5][r]*inv96);
#pragma unroll
      for (int nt = 0; nt < 4; ++nt){
        const float val = silu(accS[nt][r]*inv96) * invavg;
        atomicAdd(&aggs[(size_t)d*NS + nt*16 + lc], val);
      }
#pragma unroll
      for (int mt = 0; mt < 3; ++mt){
#pragma unroll
        for (int nt = 0; nt < 2; ++nt){
          const float g = nt ? g1 : g0;
          const float val = accV[mt*2+nt][r]*inv96*g*invavg;
          atomicAdd(&aggv[(size_t)d*(NV*3) + (nt*16+lc)*3 + mt], val);
        }
      }
    }
  }
}

// ---------------- K3: per-node update (UVU TP with attrs + linear_3) ----------------
__global__ void k_update(const float* __restrict__ aggs, const float* __restrict__ aggv,
                         const float* __restrict__ attrs,
                         const float* __restrict__ Wt0, const float* __restrict__ Wt1,
                         const float* __restrict__ W3s, const float* __restrict__ W3v,
                         float* __restrict__ flin_s, float* __restrict__ flin_v, int N){
  const int n = blockIdx.x*blockDim.x + threadIdx.x;
  if (n >= N) return;
  float at[AC];
#pragma unroll
  for (int a = 0; a < AC; ++a) at[a] = attrs[(size_t)n*AC + a];
  float q[NS];
  for (int u = 0; u < NS; ++u){
    float d0 = 0.f;
#pragma unroll
    for (int a = 0; a < AC; ++a) d0 += at[a]*Wt0[u*AC + a];
    q[u] = aggs[(size_t)n*NS + u] * d0 * 0.25f;  // /sqrt(16)
  }
  for (int v = 0; v < GS; ++v){
    float acc = 0.f;
#pragma unroll 8
    for (int u = 0; u < NS; ++u) acc += q[u]*W3s[u*GS + v];
    flin_s[(size_t)n*GS + v] = acc * 0.125f;     // /sqrt(64)
  }
  float av[NV*3];
  for (int u = 0; u < NV; ++u){
    float d1 = 0.f;
#pragma unroll
    for (int a = 0; a < AC; ++a) d1 += at[a]*Wt1[u*AC + a];
    d1 *= 0.25f;
#pragma unroll
    for (int c = 0; c < 3; ++c) av[u*3+c] = aggv[(size_t)n*(NV*3) + u*3 + c] * d1;
  }
  for (int v = 0; v < NV; ++v){
#pragma unroll
    for (int c = 0; c < 3; ++c){
      float acc = 0.f;
#pragma unroll 8
      for (int u = 0; u < NV; ++u) acc += av[u*3+c]*W3v[u*NV + v];
      flin_v[(size_t)n*(NV*3) + v*3 + c] = acc * 0.17677669529663689f; // /sqrt(32)
    }
  }
}

// ---------------- KSC: self-connection via WMMA GEMM over K = outer(node_s, attrs) ----
__global__ __launch_bounds__(128) void k_selfconn(
    const float* __restrict__ node_s, const float* __restrict__ node_v,
    const float* __restrict__ attrs,
    const float* __restrict__ Wsc0, const float* __restrict__ Wsc1,
    float* __restrict__ flin_s, float* __restrict__ flin_v, int N)
{
  __shared__ __bf16 sB0[GS*32];     // chunk of Wsc0 transposed [v][kk]
  __shared__ __bf16 sB1[NV*32];     // chunk of Wsc1 transposed [w][kk]
  __shared__ float  sS[4][16*NS];
  __shared__ float  sA[4][16*AC];
  __shared__ float  sV[4][16*NV*3];

  const int tid = threadIdx.x, w = tid >> 5, lane = tid & 31;
  const int tile = blockIdx.x*4 + w;
  const int nTiles = N/16;
  const bool tv = tile < nTiles;
  const int n0 = tile*16;
  if (tv){
    for (int i = lane; i < 16*NS; i += 32){ int m = i >> 6, u = i & 63; sS[w][i] = node_s[(size_t)(n0+m)*NS + u]; }
    for (int i = lane; i < 16*AC; i += 32){ int m = i >> 4, a = i & 15; sA[w][i] = attrs[(size_t)(n0+m)*AC + a]; }
    for (int i = lane; i < 16*96; i += 32){ int m = i/96, rr = i - m*96; sV[w][i] = node_v[(size_t)(n0+m)*96 + rr]; }
  }
  f32x8 accS[6] = {};
  f32x8 accV[6] = {};
  const int m = lane & 15, half = lane >> 4;

  for (int kt = 0; kt < 32; ++kt){
    __syncthreads();
    for (int i = tid; i < GS*32; i += 128){
      int v = i >> 5, kk = i & 31;
      sB0[i] = (__bf16)Wsc0[(size_t)(kt*32 + kk)*GS + v];
    }
    if (kt < 16){
      for (int i = tid; i < NV*32; i += 128){
        int v = i >> 5, kk = i & 31;
        sB1[i] = (__bf16)Wsc1[(size_t)(kt*32 + kk)*NV + v];
      }
    }
    __syncthreads();
    if (!tv) continue;

    bf16x16 a;
#pragma unroll
    for (int j = 0; j < 16; ++j){
      const int kl = (j < 8) ? (half*8 + j) : (16 + half*8 + (j - 8));
      const int k  = kt*32 + kl;
      a[j] = (__bf16)( sS[w][m*NS + (k >> 4)] * sA[w][m*AC + (k & 15)] );
    }
#pragma unroll
    for (int nt = 0; nt < 6; ++nt)
      accS[nt] = WMMA_BF16(a, frag_b(sB0, 32, nt, 0, lane), accS[nt]);

    if (kt < 16){
#pragma unroll
      for (int mt = 0; mt < 3; ++mt){
        bf16x16 av;
#pragma unroll
        for (int j = 0; j < 16; ++j){
          const int kl = (j < 8) ? (half*8 + j) : (16 + half*8 + (j - 8));
          const int k  = kt*32 + kl;
          av[j] = (__bf16)( sV[w][m*96 + (k >> 4)*3 + mt] * sA[w][m*AC + (k & 15)] );
        }
#pragma unroll
        for (int nt = 0; nt < 2; ++nt)
          accV[mt*2+nt] = WMMA_BF16(av, frag_b(sB1, 32, nt, 0, lane), accV[mt*2+nt]);
      }
    }
  }

  if (tv){
    const int lc = lane & 15;
#pragma unroll
    for (int r = 0; r < 8; ++r){
      const int node = n0 + r + 8*half;
#pragma unroll
      for (int nt = 0; nt < 6; ++nt)
        flin_s[(size_t)node*GS + nt*16 + lc] += accS[nt][r] * 0.03125f;           // /sqrt(1024)
#pragma unroll
      for (int mt = 0; mt < 3; ++mt)
#pragma unroll
        for (int nt = 0; nt < 2; ++nt)
          flin_v[(size_t)node*96 + (nt*16+lc)*3 + mt] += accV[mt*2+nt][r] * 0.04419417382415922f; // /sqrt(512)
    }
  }
}

// ---------------- K4: final gate + pack output [N,160] ------------------------------
__global__ void k_gate(const float* __restrict__ fs, const float* __restrict__ fv,
                       float* __restrict__ out, int N){
  const long i = (long)blockIdx.x*blockDim.x + threadIdx.x;
  const long tot = (long)N*160;
  if (i >= tot) return;
  const int n = (int)(i/160), j = (int)(i%160);
  if (j < 64) out[i] = silu(fs[(size_t)n*GS + j]);
  else {
    const int jj = j - 64, u = jj/3;
    out[i] = fv[(size_t)n*96 + jj] * sigm(fs[(size_t)n*GS + 64 + u]);
  }
}

extern "C" void kernel_launch(void* const* d_in, const int* in_sizes, int n_in,
                              void* d_out, int out_size, void* d_ws, size_t ws_size,
                              hipStream_t stream){
  const float* node_s   = (const float*)d_in[0];
  const float* node_v   = (const float*)d_in[1];
  const float* attrs    = (const float*)d_in[2];
  const float* edge_emb = (const float*)d_in[3];
  const float* edge_sh0 = (const float*)d_in[4];
  const float* edge_sh1 = (const float*)d_in[5];
  const int*   eidx     = (const int*)  d_in[6];
  const float* W1s  = (const float*)d_in[7];
  const float* W1v  = (const float*)d_in[8];
  const float* Wm1  = (const float*)d_in[9];
  const float* Wm2  = (const float*)d_in[10];
  const float* W2s  = (const float*)d_in[11];
  const float* W2v  = (const float*)d_in[12];
  const float* Wt0  = (const float*)d_in[13];
  const float* Wt1  = (const float*)d_in[14];
  const float* W3s  = (const float*)d_in[15];
  const float* W3v  = (const float*)d_in[16];
  const float* Wsc0 = (const float*)d_in[17];
  const float* Wsc1 = (const float*)d_in[18];

  const int N = in_sizes[0] / 64;
  const int E = in_sizes[3] / 8;

  char* ws = (char*)d_ws;
  float* s1     = (float*)ws; ws += (size_t)N*64*sizeof(float);
  float* v1     = (float*)ws; ws += (size_t)N*96*sizeof(float);
  float* aggs   = (float*)ws; ws += (size_t)N*64*sizeof(float);
  float* aggv   = (float*)ws; ws += (size_t)N*96*sizeof(float);
  float* flin_s = (float*)ws; ws += (size_t)N*96*sizeof(float);
  float* flin_v = (float*)ws; ws += (size_t)N*96*sizeof(float);
  float* out = (float*)d_out;

  hipMemsetAsync(aggs, 0, (size_t)N*160*sizeof(float), stream);   // aggs+aggv contiguous

  k_node_prep<<<(N+3)/4, 256, 0, stream>>>(node_s, node_v, W1s, W1v, s1, v1, N);

  const int nTiles = (E + 15) / 16;
  k_edge<<<(nTiles+1)/2, 64, 0, stream>>>(s1, v1, edge_emb, edge_sh0, edge_sh1, eidx,
                                          Wm1, Wm2, W2s, W2v, aggs, aggv, E);

  k_update<<<(N+127)/128, 128, 0, stream>>>(aggs, aggv, attrs, Wt0, Wt1, W3s, W3v,
                                            flin_s, flin_v, N);

  const int nNT = N/16;
  k_selfconn<<<(nNT+3)/4, 128, 0, stream>>>(node_s, node_v, attrs, Wsc0, Wsc1,
                                            flin_s, flin_v, N);

  const long tot = (long)N*160;
  k_gate<<<(int)((tot+255)/256), 256, 0, stream>>>(flin_s, flin_v, out, N);
}
